// LSA_15977278341170
// MI455X (gfx1250) — matrix operand order repaired and checked by reading, not compiled, not run on hardware
//
#include <hip/hip_runtime.h>
#include <hip/hip_bf16.h>
#include <math.h>

// ---------------------------------------------------------------------------
// Multi-head self-attention (B=8, N=1024, DIM=512, H=8, DH=64) for gfx1250.
// bf16 WMMA (v_wmma_f32_16x16x32_bf16) everywhere, fp32 softmax statistics.
// Weights pre-transposed+converted to bf16 (B-fragments = contiguous 32B
// loads); projection GEMMs register-blocked 2x4 (32x64 per wave) so each
// k-step runs 8 back-to-back WMMAs off one load batch.
// ---------------------------------------------------------------------------

typedef __attribute__((ext_vector_type(8)))  float   f32x8;
typedef __attribute__((ext_vector_type(8)))  __bf16  bf16x8;
typedef __attribute__((ext_vector_type(16))) __bf16  bf16x16;
typedef __attribute__((ext_vector_type(8)))  float   v8f;

#define WMMA_BF16(a, b, c) \
  __builtin_amdgcn_wmma_f32_16x16x32_bf16(false, (a), false, (b), (short)0, (c), false, false)

constexpr int Bb   = 8;
constexpr int Nn   = 1024;
constexpr int DIMD = 512;   // == H*DH
constexpr int Hh   = 8;
constexpr int DHd  = 64;
constexpr float NEG_MAX = -3.402823466e38f;

__device__ __forceinline__ float redmax16(float v) {
  #pragma unroll
  for (int m = 1; m < 16; m <<= 1) v = fmaxf(v, __shfl_xor(v, m, 32));
  return v;
}
__device__ __forceinline__ float redsum16(float v) {
  #pragma unroll
  for (int m = 1; m < 16; m <<= 1) v += __shfl_xor(v, m, 32);
  return v;
}

// ---------------------------------------------------------------------------
// Kernel 0: weight transpose + fp32->bf16 convert.  W[k][n] -> WT[n][k] bf16.
// LDS-tiled 32x32 so both global read (fp32) and write (bf16) are coalesced.
// z selects Wq / Wk / Wv / Wo.
// ---------------------------------------------------------------------------
__global__ __launch_bounds__(256) void lsa_wt_cvt(
    const float* __restrict__ Wq, const float* __restrict__ Wk,
    const float* __restrict__ Wv, const float* __restrict__ Wo,
    __bf16* __restrict__ WqT, __bf16* __restrict__ WkT,
    __bf16* __restrict__ WvT, __bf16* __restrict__ WoT)
{
  __shared__ __bf16 tile[32][33];      // +1 pad to dodge bank conflicts
  const int z = blockIdx.z;
  const float* W = (z == 0) ? Wq : (z == 1) ? Wk : (z == 2) ? Wv : Wo;
  __bf16*     WT = (z == 0) ? WqT : (z == 1) ? WkT : (z == 2) ? WvT : WoT;

  const int tx = threadIdx.x & 31;
  const int ty = threadIdx.x >> 5;     // 0..7
  const int k0 = blockIdx.x * 32;
  const int n0 = blockIdx.y * 32;

  #pragma unroll
  for (int j = 0; j < 4; ++j)
    tile[ty + 8 * j][tx] = (__bf16)W[(size_t)(k0 + ty + 8 * j) * DIMD + n0 + tx];
  __syncthreads();
  #pragma unroll
  for (int j = 0; j < 4; ++j)
    WT[(size_t)(n0 + ty + 8 * j) * DIMD + k0 + tx] = tile[tx][ty + 8 * j];
}

// ---------------------------------------------------------------------------
// Kernel 1: fused QKV projection.  X[8192,512] fp32 x WT[512,512] bf16.
// One wave computes a 32x64 strip: 2 A-fragments x 4 B-fragments per k-step
// feed 8 consecutive WMMAs (B shared across both A tiles).
// z = 0 -> Q [b,h,n,d] ; z = 1 -> K [b,h,n,d] ; z = 2 -> V^T [b,h,d,n].
// ---------------------------------------------------------------------------
__global__ __launch_bounds__(256) void lsa_qkv_proj(
    const float* __restrict__ x,
    const __bf16* __restrict__ WqT, const __bf16* __restrict__ WkT,
    const __bf16* __restrict__ WvT,
    __bf16* __restrict__ Q, __bf16* __restrict__ K, __bf16* __restrict__ Vt)
{
  const int lane = threadIdx.x & 31;
  const int wave = threadIdx.x >> 5;
  const int tid  = blockIdx.x * 8 + wave;   // 0..2047 wave-tasks
  const int tm2  = tid >> 3;                // 32-row block (256 of them)
  const int tng  = tid & 7;                 // 64-col group (8 of them)
  const int z    = blockIdx.z;
  const __bf16* WT = (z == 0) ? WqT : (z == 1) ? WkT : WvT;

  const int hi  = lane >> 4;
  const int lo  = lane & 15;
  const int kbA = hi * 16;                  // A-fragment k base
  const int kbB = hi * 16;                  // B-fragment k base
  const int col0 = tng * 64;
  const int row0 = tm2 * 32;
  const float* xr0 = x + (size_t)(row0 + lo) * DIMD;        // A tile 0 row
  const float* xr1 = x + (size_t)(row0 + 16 + lo) * DIMD;   // A tile 1 row

  v8f acc[2][4] = {};
  for (int k0 = 0; k0 < DIMD; k0 += 32) {
    // A fragments for both M tiles (fp32 -> bf16 once, reused by 4 WMMAs each)
    f32x8 a0lo = *(const f32x8*)(xr0 + k0 + kbA);
    f32x8 a0hi = *(const f32x8*)(xr0 + k0 + kbA + 8);
    f32x8 a1lo = *(const f32x8*)(xr1 + k0 + kbA);
    f32x8 a1hi = *(const f32x8*)(xr1 + k0 + kbA + 8);
    bf16x16 av0, av1;
    #pragma unroll
    for (int e = 0; e < 8; ++e) {
      av0[e] = (__bf16)a0lo[e]; av0[8 + e] = (__bf16)a0hi[e];
      av1[e] = (__bf16)a1lo[e]; av1[8 + e] = (__bf16)a1hi[e];
    }
    // B fragments: WT[n][k] -> 16 contiguous bf16 per lane (32B load)
    bf16x16 bv[4];
    #pragma unroll
    for (int t = 0; t < 4; ++t)
      bv[t] = *(const bf16x16*)(WT + (size_t)(col0 + t * 16 + lo) * DIMD + k0 + kbB);
    // 8 back-to-back WMMAs
    #pragma unroll
    for (int t = 0; t < 4; ++t) {
      acc[0][t] = WMMA_BF16(av0, bv[t], acc[0][t]);
      acc[1][t] = WMMA_BF16(av1, bv[t], acc[1][t]);
    }
  }

  // Scatter C tiles (row = r + 8*hi, col = lane&15) into head-major layouts.
  #pragma unroll
  for (int mt = 0; mt < 2; ++mt) {
    #pragma unroll
    for (int t = 0; t < 4; ++t) {
      const int col = col0 + t * 16 + lo;
      const int h = col >> 6, d = col & 63;
      #pragma unroll
      for (int r = 0; r < 8; ++r) {
        const int row = row0 + mt * 16 + r + 8 * hi;
        const int b = row >> 10, n = row & 1023;
        const __bf16 v = (__bf16)acc[mt][t][r];
        if (z == 2) {
          Vt[(((size_t)(b * Hh + h)) * DHd + d) * Nn + n] = v;   // V transposed
        } else {
          __bf16* O = (z == 0) ? Q : K;
          O[(((size_t)(b * Hh + h)) * Nn + n) * DHd + d] = v;
        }
      }
    }
  }
}

// ---------------------------------------------------------------------------
// Kernel 2: flash attention.  4 waves/block, each wave owns 16 query rows of
// one (b,h); streams 32-key blocks: 4 QK^T WMMAs + softmax update + 4 PV WMMAs.
// V B-fragments are loaded BEFORE the softmax VALU block so the loads are in
// flight during the shuffle/exp work and the LDS P round-trip.
// ---------------------------------------------------------------------------
__global__ __launch_bounds__(128) void lsa_flash_attn(
    const __bf16* __restrict__ Q, const __bf16* __restrict__ K,
    const __bf16* __restrict__ Vt, const float* __restrict__ log_temp,
    __bf16* __restrict__ AO)
{
  __shared__ __align__(32) __bf16 plds[4][16 * 32];

  const int lane = threadIdx.x & 31;
  const int wave = threadIdx.x >> 5;
  const int bh   = blockIdx.y;              // 0..63
  const int b    = bh >> 3, h = bh & 7;
  const int qtile = blockIdx.x * 64 + wave * 16;
  const int hi = lane >> 4, lo = lane & 15;
  const float temp = __expf(log_temp[0]);

  const __bf16* Qh = Q  + (size_t)bh * Nn * DHd;
  const __bf16* Kh = K  + (size_t)bh * Nn * DHd;
  const __bf16* Vh = Vt + (size_t)bh * DHd * Nn;

  const int kbA = hi * 16;
  const int kbB = hi * 16;

  // Q A-fragments for k = 0..31 and 32..63 (resident for the whole pass)
  bf16x16 aq0, aq1;
  {
    const __bf16* qr = Qh + (size_t)(qtile + lo) * DHd;
    bf16x8 q00 = *(const bf16x8*)(qr + kbA);
    bf16x8 q01 = *(const bf16x8*)(qr + kbA + 8);
    bf16x8 q10 = *(const bf16x8*)(qr + 32 + kbA);
    bf16x8 q11 = *(const bf16x8*)(qr + 32 + kbA + 8);
    #pragma unroll
    for (int e = 0; e < 8; ++e) {
      aq0[e] = q00[e]; aq0[8 + e] = q01[e];
      aq1[e] = q10[e]; aq1[8 + e] = q11[e];
    }
  }

  float m[8], lsum[8];
  #pragma unroll
  for (int r = 0; r < 8; ++r) { m[r] = NEG_MAX; lsum[r] = 0.0f; }
  v8f oacc[4] = {};

  for (int j0 = 0; j0 < Nn; j0 += 32) {
    if (j0 + 32 < Nn)
      __builtin_prefetch(Kh + (size_t)(j0 + 32) * DHd + lane * 16, 0, 1);

    // ---- scores: two 16x16 tiles covering keys [j0, j0+32) ----
    v8f s[2];
    #pragma unroll
    for (int u = 0; u < 2; ++u) {
      const __bf16* kr = Kh + (size_t)(j0 + u * 16 + lo) * DHd;
      bf16x16 bk0 = *(const bf16x16*)(kr + kbB);
      bf16x16 bk1 = *(const bf16x16*)(kr + 32 + kbB);
      v8f sz = {};
      sz = WMMA_BF16(aq0, bk0, sz);
      sz = WMMA_BF16(aq1, bk1, sz);
      s[u] = sz;
    }

    // ---- V B-fragments: issue loads NOW, consume after softmax ----
    bf16x16 bv[4];
    #pragma unroll
    for (int t = 0; t < 4; ++t)
      bv[t] = *(const bf16x16*)(Vh + (size_t)(t * 16 + lo) * Nn + j0 + kbB);

    // ---- online softmax update (fp32) ----
    #pragma unroll
    for (int r = 0; r < 8; ++r) {
      const int qg = qtile + r + 8 * hi;
      float sa = s[0][r] * temp; if (qg == j0 + lo)      sa = NEG_MAX;
      float sb = s[1][r] * temp; if (qg == j0 + 16 + lo) sb = NEG_MAX;
      const float rm   = redmax16(fmaxf(sa, sb));
      const float mn   = fmaxf(m[r], rm);
      const float corr = __expf(m[r] - mn);
      const float pa = __expf(sa - mn);
      const float pb = __expf(sb - mn);
      lsum[r] = lsum[r] * corr + redsum16(pa + pb);
      m[r] = mn;
      oacc[0][r] *= corr; oacc[1][r] *= corr;
      oacc[2][r] *= corr; oacc[3][r] *= corr;
      // stage P (C-layout) into LDS as bf16 [query][key_in_block]
      plds[wave][(r + 8 * hi) * 32 + lo]      = (__bf16)pa;
      plds[wave][(r + 8 * hi) * 32 + 16 + lo] = (__bf16)pb;
    }

    asm volatile("s_wait_dscnt 0" ::: "memory");

    // ---- reload P in A-fragment layout (16 x 32) ----
    bf16x16 ap;
    {
      bf16x8 p0 = *(const bf16x8*)(&plds[wave][lo * 32 + kbA]);
      bf16x8 p1 = *(const bf16x8*)(&plds[wave][lo * 32 + kbA + 8]);
      #pragma unroll
      for (int e = 0; e < 8; ++e) { ap[e] = p0[e]; ap[8 + e] = p1[e]; }
    }

    // ---- O += P @ V (V loads long since landed) ----
    #pragma unroll
    for (int t = 0; t < 4; ++t)
      oacc[t] = WMMA_BF16(ap, bv[t], oacc[t]);
  }

  // ---- normalize and store AO[b, n, h*64 + d] as bf16 ----
  #pragma unroll
  for (int t = 0; t < 4; ++t) {
    #pragma unroll
    for (int r = 0; r < 8; ++r) {
      const int qg = qtile + r + 8 * hi;
      const float v = oacc[t][r] / lsum[r];
      AO[((size_t)(b * Nn + qg)) * DIMD + h * DHd + t * 16 + lo] = (__bf16)v;
    }
  }
}

// ---------------------------------------------------------------------------
// Kernel 3: output projection.  AO[8192,512] bf16 x WoT bf16 + bo -> fp32.
// Same 2x4 register blocking; zero conversions in the inner loop.
// ---------------------------------------------------------------------------
__global__ __launch_bounds__(256) void lsa_out_proj(
    const __bf16* __restrict__ AO, const __bf16* __restrict__ WoT,
    const float* __restrict__ bo, float* __restrict__ out)
{
  const int lane = threadIdx.x & 31;
  const int wave = threadIdx.x >> 5;
  const int tid  = blockIdx.x * 8 + wave;   // 0..2047
  const int tm2  = tid >> 3;
  const int tng  = tid & 7;
  const int hi = lane >> 4, lo = lane & 15;
  const int row0 = tm2 * 32;
  const int kbA = hi * 16;
  const int kbB = hi * 16;
  const int col0 = tng * 64;
  const __bf16* ar0 = AO + (size_t)(row0 + lo) * DIMD;
  const __bf16* ar1 = AO + (size_t)(row0 + 16 + lo) * DIMD;

  v8f acc[2][4] = {};
  for (int k0 = 0; k0 < DIMD; k0 += 32) {
    bf16x8 a00 = *(const bf16x8*)(ar0 + k0 + kbA);
    bf16x8 a01 = *(const bf16x8*)(ar0 + k0 + kbA + 8);
    bf16x8 a10 = *(const bf16x8*)(ar1 + k0 + kbA);
    bf16x8 a11 = *(const bf16x8*)(ar1 + k0 + kbA + 8);
    bf16x16 av0, av1;
    #pragma unroll
    for (int e = 0; e < 8; ++e) {
      av0[e] = a00[e]; av0[8 + e] = a01[e];
      av1[e] = a10[e]; av1[8 + e] = a11[e];
    }
    bf16x16 bv[4];
    #pragma unroll
    for (int t = 0; t < 4; ++t)
      bv[t] = *(const bf16x16*)(WoT + (size_t)(col0 + t * 16 + lo) * DIMD + k0 + kbB);
    #pragma unroll
    for (int t = 0; t < 4; ++t) {
      acc[0][t] = WMMA_BF16(av0, bv[t], acc[0][t]);
      acc[1][t] = WMMA_BF16(av1, bv[t], acc[1][t]);
    }
  }

  #pragma unroll
  for (int mt = 0; mt < 2; ++mt) {
    #pragma unroll
    for (int t = 0; t < 4; ++t) {
      const int col = col0 + t * 16 + lo;
      const float bias = bo[col];
      #pragma unroll
      for (int r = 0; r < 8; ++r)
        out[(size_t)(row0 + mt * 16 + r + 8 * hi) * DIMD + col] = acc[mt][t][r] + bias;
    }
  }
}

// ---------------------------------------------------------------------------
// Launch: wt-convert -> qkv -> flash attention -> output projection.
// Workspace: Q | K | V^T | AO (4 x 8 MiB bf16) + 4 transposed bf16 weights
// (4 x 512 KiB) = 34 MiB.
// ---------------------------------------------------------------------------
extern "C" void kernel_launch(void* const* d_in, const int* in_sizes, int n_in,
                              void* d_out, int out_size, void* d_ws, size_t ws_size,
                              hipStream_t stream) {
  const float* x        = (const float*)d_in[0];
  const float* Wq       = (const float*)d_in[1];
  const float* Wk       = (const float*)d_in[2];
  const float* Wv       = (const float*)d_in[3];
  const float* Wo       = (const float*)d_in[4];
  const float* bo       = (const float*)d_in[5];
  const float* log_temp = (const float*)d_in[6];
  float* out = (float*)d_out;

  const size_t TOK = (size_t)Bb * Nn * DIMD;   // 8192 * 512 elements
  const size_t WEL = (size_t)DIMD * DIMD;      // 512 * 512 elements
  __bf16* Q   = (__bf16*)d_ws;
  __bf16* K   = Q   + TOK;
  __bf16* Vt  = K   + TOK;
  __bf16* AO  = Vt  + TOK;
  __bf16* WqT = AO  + TOK;
  __bf16* WkT = WqT + WEL;
  __bf16* WvT = WkT + WEL;
  __bf16* WoT = WvT + WEL;

  // transpose+convert all four weight matrices to bf16 [n][k]
  lsa_wt_cvt<<<dim3(16, 16, 4), 256, 0, stream>>>(Wq, Wk, Wv, Wo,
                                                  WqT, WkT, WvT, WoT);
  // 2048 wave-tasks (32x64 strips) / 8 waves per block, z selects Q/K/V
  lsa_qkv_proj<<<dim3(256, 1, 3), 256, 0, stream>>>(x, WqT, WkT, WvT, Q, K, Vt);
  // 64 (b,h) pairs x 16 query blocks of 64 rows
  lsa_flash_attn<<<dim3(16, 64), 128, 0, stream>>>(Q, K, Vt, log_temp, AO);
  // final projection
  lsa_out_proj<<<dim3(256), 256, 0, stream>>>(AO, WoT, bo, out);
}
